// CotrackerIter_4621384810705
// MI455X (gfx1250) — matrix operand order repaired and checked by reading, not compiled, not run on hardware
//
// CotrackerIter for MI455X (gfx1250, CDNA5, wave32).
// All dense GEMMs -> v_wmma_f32_16x16x32_bf16 (f32 in -> bf16 fragments, f32 acc).
// 128x128 workgroup tile, 8 waves x (32x64) sub-tiles -> 8 WMMAs per K-step,
// double-buffered LDS staging with vectorized (float4) interior fast path and
// guarded scalar edge path; fused bias+ReLU epilogue; A-stream prefetch.
// Attention uses wave32 shfl reductions (hd = 32 = wave); MSDA sampling: lane = channel; LN: LDS tree sum.
// msda 'value' projection is hoisted out of the refinement loop (src is loop-invariant).
// NOTE: workspace layout needs ~250 MB; input pointer order assumes JAX pytree
// flattening (dict keys sorted) of setup_inputs().

#include <hip/hip_runtime.h>
#include <hip/hip_bf16.h>
#include <math.h>

typedef __bf16 bf16_t;
typedef __attribute__((ext_vector_type(16))) __bf16 v16bf;
typedef __attribute__((ext_vector_type(8)))  __bf16 v8bf;
typedef __attribute__((ext_vector_type(4)))  __bf16 v4bf;
typedef __attribute__((ext_vector_type(8)))  float  v8f;

#define GTM 128
#define GTN 128
#define GTK 32

// ---------------------------------------------------------------------------
// Stage one (A 128x32, B 32x128) tile pair into LDS as bf16.
// GUARD=false: fully-interior tile, float4 vector loads (b128), packed stores.
// ---------------------------------------------------------------------------
template <bool GUARD>
__device__ __forceinline__ void stage_tile(
    const float* __restrict__ A, const float* __restrict__ B,
    bf16_t* __restrict__ sA, bf16_t* __restrict__ sB,
    int M, int N, int K, int tileM, int tileN, int k0, int tid)
{
    // A tile: GTM x GTK, vec4 along K. 1024 float4 -> 4 per thread.
    #pragma unroll
    for (int j = 0; j < (GTM * GTK / 4) / 256; ++j) {
        int idx = tid + j * 256;
        int r  = idx >> 3;
        int c4 = (idx & 7) << 2;
        int gm = tileM + r, gk = k0 + c4;
        float4 v;
        if (!GUARD) {
            v = *(const float4*)(A + (size_t)gm * K + gk);
            if (gk + 2 * GTK < K)                       // stream 2 K-tiles ahead
                __builtin_prefetch(A + (size_t)gm * K + gk + 2 * GTK, 0, 1);
        } else {
            v.x = (gm < M && gk + 0 < K) ? A[(size_t)gm * K + gk + 0] : 0.f;
            v.y = (gm < M && gk + 1 < K) ? A[(size_t)gm * K + gk + 1] : 0.f;
            v.z = (gm < M && gk + 2 < K) ? A[(size_t)gm * K + gk + 2] : 0.f;
            v.w = (gm < M && gk + 3 < K) ? A[(size_t)gm * K + gk + 3] : 0.f;
        }
        v4bf pk = { (bf16_t)v.x, (bf16_t)v.y, (bf16_t)v.z, (bf16_t)v.w };
        *(v4bf*)(sA + r * GTK + c4) = pk;
    }
    // B tile: GTK x GTN in memory (row n-contiguous), stored to LDS as [n][k].
    #pragma unroll
    for (int j = 0; j < (GTK * GTN / 4) / 256; ++j) {
        int idx = tid + j * 256;
        int k  = idx >> 5;
        int n4 = (idx & 31) << 2;
        int gk = k0 + k, gn = tileN + n4;
        float4 v;
        if (!GUARD) {
            v = *(const float4*)(B + (size_t)gk * N + gn);
        } else {
            bool kin = gk < K;
            v.x = (kin && gn + 0 < N) ? B[(size_t)gk * N + gn + 0] : 0.f;
            v.y = (kin && gn + 1 < N) ? B[(size_t)gk * N + gn + 1] : 0.f;
            v.z = (kin && gn + 2 < N) ? B[(size_t)gk * N + gn + 2] : 0.f;
            v.w = (kin && gn + 3 < N) ? B[(size_t)gk * N + gn + 3] : 0.f;
        }
        sB[(n4 + 0) * GTK + k] = (bf16_t)v.x;
        sB[(n4 + 1) * GTK + k] = (bf16_t)v.y;
        sB[(n4 + 2) * GTK + k] = (bf16_t)v.z;
        sB[(n4 + 3) * GTK + k] = (bf16_t)v.w;
    }
}

// ---------------------------------------------------------------------------
// WMMA GEMM: C[M,N] = act(A[M,K] * B[K,N] + bias[N]);  act: 0=none, 1=relu
// ---------------------------------------------------------------------------
__global__ __launch_bounds__(256) void gemm_bias_act_wmma_kernel(
    const float* __restrict__ A, const float* __restrict__ B,
    const float* __restrict__ bias, float* __restrict__ C,
    int M, int N, int K, int act)
{
    __shared__ __align__(16) bf16_t sA[2][GTM * GTK];
    __shared__ __align__(16) bf16_t sB[2][GTN * GTK];

    const int tid  = threadIdx.x;
    const int lane = tid & 31;
    const int wid  = tid >> 5;          // 0..7
    const int wm   = wid & 3;           // wave row   (4 x 32 = 128 rows)
    const int wn   = wid >> 2;          // wave col   (2 x 64 = 128 cols)
    const int half = lane >> 4;         // 0 | 1
    const int l16  = lane & 15;
    const int tileM = blockIdx.y * GTM;
    const int tileN = blockIdx.x * GTN;
    const bool interior = (tileM + GTM <= M) && (tileN + GTN <= N);

    v8f acc[2][4];
    for (int s = 0; s < 2; ++s)
        for (int t = 0; t < 4; ++t)
            for (int e = 0; e < 8; ++e) acc[s][t][e] = 0.f;

    // prologue: stage first tile into buffer 0
    if (interior && GTK <= K) stage_tile<false>(A, B, sA[0], sB[0], M, N, K, tileM, tileN, 0, tid);
    else                      stage_tile<true >(A, B, sA[0], sB[0], M, N, K, tileM, tileN, 0, tid);
    __syncthreads();

    int par = 0;
    for (int k0 = 0; k0 < K; k0 += GTK) {
        // stage next K-tile into the other buffer while computing this one
        int kn = k0 + GTK;
        if (kn < K) {
            if (interior && kn + GTK <= K)
                stage_tile<false>(A, B, sA[par ^ 1], sB[par ^ 1], M, N, K, tileM, tileN, kn, tid);
            else
                stage_tile<true >(A, B, sA[par ^ 1], sB[par ^ 1], M, N, K, tileM, tileN, kn, tid);
        }

        const bf16_t* cA = sA[par];
        const bf16_t* cB = sB[par];

        // A fragments for the wave's two 16-row sub-tiles
        // (ISA 7.12.2: lane<16 -> K 0..7,16..23 ; lane>=16 -> +8)
        v16bf af[2];
        #pragma unroll
        for (int s = 0; s < 2; ++s) {
            const bf16_t* ap = cA + (wm * 32 + s * 16 + l16) * GTK;
            v8bf lo = *(const v8bf*)(ap + 8 * half);
            v8bf hi = *(const v8bf*)(ap + 16 + 8 * half);
            #pragma unroll
            for (int e = 0; e < 8; ++e) { af[s][e] = lo[e]; af[s][e + 8] = hi[e]; }
        }
        // B fragment per 16-col sub-tile, reused by both M sub-tiles
        #pragma unroll
        for (int t = 0; t < 4; ++t) {
            const bf16_t* bp = cB + (wn * 64 + t * 16 + l16) * GTK + 8 * half;
            v8bf b0 = *(const v8bf*)bp;
            v8bf b1 = *(const v8bf*)(bp + 16);
            v16bf bfr;
            #pragma unroll
            for (int e = 0; e < 8; ++e) { bfr[e] = b0[e]; bfr[e + 8] = b1[e]; }
            acc[0][t] = __builtin_amdgcn_wmma_f32_16x16x32_bf16(
                false, af[0], false, bfr, (short)0, acc[0][t], false, false);
            acc[1][t] = __builtin_amdgcn_wmma_f32_16x16x32_bf16(
                false, af[1], false, bfr, (short)0, acc[1][t], false, false);
        }
        __syncthreads();
        par ^= 1;
    }

    // Epilogue: D layout -> row = r + 8*half, col = lane&15 (16x16 f32 C/D map)
    for (int s = 0; s < 2; ++s) {
        for (int t = 0; t < 4; ++t) {
            int col = tileN + wn * 64 + t * 16 + l16;
            float bv = (bias != nullptr && col < N) ? bias[col] : 0.f;
            for (int r = 0; r < 8; ++r) {
                int row = tileM + wm * 32 + s * 16 + r + 8 * half;
                if (row < M && col < N) {
                    float v = acc[s][t][r] + bv;
                    if (act == 1) v = fmaxf(v, 0.f);
                    C[(size_t)row * N + col] = v;
                }
            }
        }
    }
}

// ---------------------------------------------------------------------------
// LayerNorm over last dim (=256), optional residual: out = ln(x (+ res))
// ---------------------------------------------------------------------------
__global__ void ln_residual_kernel(const float* __restrict__ x,
                                   const float* __restrict__ res,
                                   float* __restrict__ out, int R)
{
    int row = blockIdx.x;
    if (row >= R) return;
    int t = threadIdx.x;                       // 256 threads
    float v = x[(size_t)row * 256 + t];
    if (res) v += res[(size_t)row * 256 + t];
    __shared__ float red[256];
    red[t] = v; __syncthreads();
    for (int s = 128; s > 0; s >>= 1) { if (t < s) red[t] += red[t + s]; __syncthreads(); }
    float mu = red[0] * (1.f / 256.f);
    __syncthreads();
    float d = v - mu;
    red[t] = d * d; __syncthreads();
    for (int s = 128; s > 0; s >>= 1) { if (t < s) red[t] += red[t + s]; __syncthreads(); }
    float var = red[0] * (1.f / 256.f);
    out[(size_t)row * 256 + t] = d * rsqrtf(var + 1e-5f);
}

// ---------------------------------------------------------------------------
// Softmax attention, hd == 32 == wave32: one wave per (b,h,q). Lk <= 32.
// qh/kh/vh: (B*L,256) with head h at cols [h*32, h*32+32)
// ---------------------------------------------------------------------------
__global__ void attn_kernel(const float* __restrict__ qh, const float* __restrict__ kh,
                            const float* __restrict__ vh, float* __restrict__ o,
                            int Lq, int Lk)
{
    int idx = blockIdx.x;
    int lane = threadIdx.x;
    int qi = idx % Lq;
    int h  = (idx / Lq) & 7;
    int b  = idx / (Lq * 8);
    const float* qrow = qh + ((size_t)(b * Lq + qi)) * 256 + h * 32;
    float score = -1e30f;
    if (lane < Lk) {
        const float* krow = kh + ((size_t)(b * Lk + lane)) * 256 + h * 32;
        float a = 0.f;
        for (int d = 0; d < 32; ++d) a += qrow[d] * krow[d];
        score = a * 0.17677669529663687f;      // 1/sqrt(32)
    }
    float m = score;
    for (int s = 16; s > 0; s >>= 1) m = fmaxf(m, __shfl_xor(m, s));
    float e = (lane < Lk) ? __expf(score - m) : 0.f;
    float sum = e;
    for (int s = 16; s > 0; s >>= 1) sum += __shfl_xor(sum, s);
    float p = e / sum;
    float outv = 0.f;
    for (int k = 0; k < Lk; ++k) {
        float pk = __shfl(p, k);
        outv += pk * vh[((size_t)(b * Lk + k)) * 256 + h * 32 + lane];
    }
    o[((size_t)(b * Lq + qi)) * 256 + h * 32 + lane] = outv;
}

// ---------------------------------------------------------------------------
// MSDA: softmax over 36 sampling weights per (b,q,h)
// aw: (B*Q, 288), head h at cols [h*36, h*36+36)
// ---------------------------------------------------------------------------
__global__ void aw_softmax_kernel(float* __restrict__ aw, int rows)
{
    int i = blockIdx.x * 64 + threadIdx.x;
    if (i >= rows * 8) return;
    float* a = aw + (size_t)(i >> 3) * 288 + (i & 7) * 36;
    float m = a[0];
    for (int k = 1; k < 36; ++k) m = fmaxf(m, a[k]);
    float s = 0.f;
    for (int k = 0; k < 36; ++k) { float e = __expf(a[k] - m); a[k] = e; s += e; }
    float inv = 1.f / s;
    for (int k = 0; k < 36; ++k) a[k] *= inv;
}

// ---------------------------------------------------------------------------
// MSDA bilinear sampling + weighted sum: one wave per (b,q,h); lane = channel
// value: (B, 29008, 256); off: (B*Q,576); aw: (B*Q,288) softmaxed; ref: (B,Q,2)
// ---------------------------------------------------------------------------
__global__ void msda_sample_kernel(const float* __restrict__ value,
                                   const float* __restrict__ off,
                                   const float* __restrict__ aw,
                                   const float* __restrict__ ref,
                                   float* __restrict__ out, int Q)
{
    int idx = blockIdx.x, lane = threadIdx.x;
    int h = idx & 7;
    int q = (idx >> 3) % Q;
    int b = idx / (8 * Q);
    const int Hs[4] = {28, 56, 112, 112};
    const int st[4] = {0, 784, 3920, 16464};
    float rx = ref[((size_t)b * Q + q) * 2 + 0];
    float ry = ref[((size_t)b * Q + q) * 2 + 1];
    const float* op = off + ((size_t)b * Q + q) * 576 + h * 72;
    const float* ap = aw  + ((size_t)b * Q + q) * 288 + h * 36;
    const float* vb = value + (size_t)b * 29008 * 256 + h * 32 + lane;
    float acc = 0.f;
    for (int l = 0; l < 4; ++l) {
        int Hl = Hs[l], Wl = Hs[l], s0 = st[l];
        for (int p = 0; p < 9; ++p) {
            float lx = rx + op[(l * 9 + p) * 2 + 0] / (float)Wl;
            float ly = ry + op[(l * 9 + p) * 2 + 1] / (float)Hl;
            float x = lx * Wl - 0.5f, y = ly * Hl - 0.5f;
            float xf = floorf(x), yf = floorf(y);
            int x0 = (int)xf, y0 = (int)yf;
            float wx = x - xf, wy = y - yf;
            float w = ap[l * 9 + p];
            float v00 = (y0 >= 0   && y0 < Hl   && x0 >= 0   && x0 < Wl)   ? vb[(size_t)(s0 + y0 * Wl + x0) * 256] : 0.f;
            float v01 = (y0 >= 0   && y0 < Hl   && x0+1 >= 0 && x0+1 < Wl) ? vb[(size_t)(s0 + y0 * Wl + x0 + 1) * 256] : 0.f;
            float v10 = (y0+1 >= 0 && y0+1 < Hl && x0 >= 0   && x0 < Wl)   ? vb[(size_t)(s0 + (y0+1) * Wl + x0) * 256] : 0.f;
            float v11 = (y0+1 >= 0 && y0+1 < Hl && x0+1 >= 0 && x0+1 < Wl) ? vb[(size_t)(s0 + (y0+1) * Wl + x0 + 1) * 256] : 0.f;
            acc += w * (v00 * (1-wx)*(1-wy) + v01 * wx*(1-wy) + v10 * (1-wx)*wy + v11 * wx*wy);
        }
    }
    out[((size_t)b * Q + q) * 256 + h * 32 + lane] = acc;
}

// ---------------------------------------------------------------------------
// (N,384,112*112) channel-first -> (N*112*112, 384) token-major
// ---------------------------------------------------------------------------
__global__ void chw2hwc_kernel(const float* __restrict__ in, float* __restrict__ out, int N)
{
    size_t i = (size_t)blockIdx.x * 256 + threadIdx.x;
    size_t total = (size_t)N * 384 * 12544;
    if (i >= total) return;
    int p = (int)(i % 12544);
    size_t t = i / 12544;
    int c = (int)(t % 384);
    int n = (int)(t / 384);
    out[((size_t)n * 12544 + p) * 384 + c] = in[i];
}

// ---------------------------------------------------------------------------
// jax.image.resize 'bilinear' (antialias) from (N,112,112,256) into src level
// ---------------------------------------------------------------------------
__global__ void resize_to_src_kernel(const float* __restrict__ xin, float* __restrict__ src,
                                     int Nimg, int Hout, int tokStart)
{
    size_t i = (size_t)blockIdx.x * 256 + threadIdx.x;
    size_t total = (size_t)Nimg * Hout * Hout * 256;
    if (i >= total) return;
    int c = (int)(i & 255);
    size_t t = i >> 8;
    int x = (int)(t % Hout); t /= Hout;
    int y = (int)(t % Hout);
    int n = (int)(t / Hout);
    float s = 112.f / (float)Hout;           // inv scale
    float supp = fmaxf(s, 1.f);
    float cy = ((float)y + 0.5f) * s - 0.5f;
    float cx = ((float)x + 0.5f) * s - 0.5f;
    int y0 = (int)ceilf(cy - supp), y1 = (int)floorf(cy + supp);
    int x0 = (int)ceilf(cx - supp), x1 = (int)floorf(cx + supp);
    float acc = 0.f, wsum = 0.f;
    for (int yy = y0; yy <= y1; ++yy) {
        float wy = 1.f - fabsf(((float)yy - cy) / supp);
        if (wy <= 0.f) continue;
        int yc = yy < 0 ? 0 : (yy > 111 ? 111 : yy);
        for (int xx = x0; xx <= x1; ++xx) {
            float wx = 1.f - fabsf(((float)xx - cx) / supp);
            if (wx <= 0.f) continue;
            int xc = xx < 0 ? 0 : (xx > 111 ? 111 : xx);
            float w = wy * wx;
            acc += w * xin[(((size_t)n * 112 + yc) * 112 + xc) * 256 + c];
            wsum += w;
        }
    }
    src[((size_t)n * 29008 + tokStart + (size_t)y * Hout + x) * 256 + c] = acc / wsum;
}

// ---------------------------------------------------------------------------
// src += pos2d(level geometry) + level_pos[level]   (token layout 28,56,112,112)
// ---------------------------------------------------------------------------
__global__ void add_pos2d_kernel(float* __restrict__ src, const float* __restrict__ level_pos,
                                 int Nimg)
{
    size_t i = (size_t)blockIdx.x * 256 + threadIdx.x;
    size_t total = (size_t)Nimg * 29008 * 256;
    if (i >= total) return;
    int c = (int)(i & 255);
    size_t t = i >> 8;
    int tok = (int)(t % 29008);
    int l, H, W, rel;
    if      (tok < 784)   { l = 0; H = 28;  W = 28;  rel = tok; }
    else if (tok < 3920)  { l = 1; H = 56;  W = 56;  rel = tok - 784; }
    else if (tok < 16464) { l = 2; H = 112; W = 112; rel = tok - 3920; }
    else                  { l = 3; H = 112; W = 112; rel = tok - 16464; }
    int y = rel / W, x = rel % W;
    float coord; int j;
    const float twopi = 6.283185307179586f;
    if (c < 128) { coord = (float)(y + 1) / ((float)H + 1e-6f) * twopi; j = c; }
    else         { coord = (float)(x + 1) / ((float)W + 1e-6f) * twopi; j = c - 128; }
    float dimt = __powf(10000.f, 2.f * (float)(j >> 1) / 128.f);
    float ph = coord / dimt;
    float pe = (j & 1) ? __cosf(ph) : __sinf(ph);
    src[i] += pe + level_pos[l * 256 + c];
}

// x[(b,l,c)] += pos1d(L)
__global__ void add_pos1d_kernel(float* __restrict__ x, int total, int L)
{
    int i = blockIdx.x * 256 + threadIdx.x;
    if (i >= total) return;
    int c = i & 255;
    int l = (i >> 8) % L;
    int k = c >> 1;
    float div = __expf((float)(2 * k) * (-9.210340371976184f / 256.f));
    float ph = (float)l * div;
    x[i] += (c & 1) ? __cosf(ph) : __sinf(ph);
}

__global__ void bcast_rows_kernel(const float* __restrict__ src, float* __restrict__ dst,
                                  int total, int P)
{
    int i = blockIdx.x * 256 + threadIdx.x;
    if (i >= total) return;
    int c = i & 255;
    int p = (i >> 8) % P;
    dst[i] = src[p * 256 + c];
}

__global__ void scale_copy_kernel(const float* __restrict__ in, float* __restrict__ out,
                                  int n, float s)
{
    int i = blockIdx.x * 256 + threadIdx.x;
    if (i < n) out[i] = in[i] * s;
}

__global__ void axpy_kernel(float* __restrict__ y, const float* __restrict__ x, int n)
{
    int i = blockIdx.x * 256 + threadIdx.x;
    if (i < n) y[i] += x[i];
}

// extreme4: first-occurrence argmin/argmax over 16 points; broadcast to 2 frames
__global__ void extreme4_kernel(const float* __restrict__ pts, float* __restrict__ outp)
{
    if (threadIdx.x != 0 || blockIdx.x != 0) return;
    int it = 0, ir = 0, ib = 0, il = 0;
    for (int i = 1; i < 16; ++i) {
        if (pts[i * 2 + 1] < pts[it * 2 + 1]) it = i;
        if (pts[i * 2 + 0] > pts[ir * 2 + 0]) ir = i;
        if (pts[i * 2 + 1] > pts[ib * 2 + 1]) ib = i;
        if (pts[i * 2 + 0] < pts[il * 2 + 0]) il = i;
    }
    int idx[4] = {it, ir, ib, il};
    for (int n = 0; n < 2; ++n)
        for (int e = 0; e < 4; ++e) {
            outp[(n * 4 + e) * 2 + 0] = pts[idx[e] * 2 + 0];
            outp[(n * 4 + e) * 2 + 1] = pts[idx[e] * 2 + 1];
        }
}

// pts += (sigmoid(mlp)-0.5)*0.3 ; outSlice = pts*224
__global__ void update_points_kernel(float* __restrict__ pts, const float* __restrict__ mlpout,
                                     float* __restrict__ outSlice, int elems)
{
    int i = blockIdx.x * 256 + threadIdx.x;
    if (i >= elems) return;
    float d = (1.f / (1.f + __expf(-mlpout[i])) - 0.5f) * 0.3f;
    float v = pts[i] + d;
    pts[i] = v;
    outSlice[i] = v * 224.f;
}

// out[2j]=(p[j]+p[j-1])/2 ; out[2j+1]=p[j]   per frame
__global__ void subdivide_points_kernel(const float* __restrict__ pin, float* __restrict__ pout,
                                        int p)
{
    int i = blockIdx.x * 256 + threadIdx.x;
    int total = 2 * p * 2;
    if (i >= total) return;
    int c = i & 1;
    int j = (i >> 1) % p;
    int n = i / (p * 2);
    float cur  = pin[(n * p + j) * 2 + c];
    float prev = pin[(n * p + ((j + p - 1) % p)) * 2 + c];
    pout[(n * 2 * p + 2 * j) * 2 + c]     = 0.5f * (cur + prev);
    pout[(n * 2 * p + 2 * j + 1) * 2 + c] = cur;
}

// out[2j]=query_embed[p+j] ; out[2j+1]=curq[j]
__global__ void interleave_q_kernel(const float* __restrict__ qe, const float* __restrict__ curq,
                                    float* __restrict__ outq, int p)
{
    int i = blockIdx.x * 256 + threadIdx.x;
    int total = 2 * (2 * p) * 256;
    if (i >= total) return;
    int c = i & 255;
    int j = (i >> 8) % (2 * p);
    int n = i / ((2 * p) * 256);
    float v;
    if ((j & 1) == 0) v = qe[(p + (j >> 1)) * 256 + c];
    else              v = curq[((size_t)n * p + (j >> 1)) * 256 + c];
    outq[i] = v;
}

// ---------------------------------------------------------------------------
struct MHAp { const float *wq,*bq,*wk,*bk,*wv,*bv,*wo,*bo; };

extern "C" void kernel_launch(void* const* d_in, const int* in_sizes, int n_in,
                              void* d_out, int out_size, void* d_ws, size_t ws_size,
                              hipStream_t stream)
{
    (void)in_sizes; (void)n_in; (void)out_size; (void)ws_size;
    auto in = [&](int i) { return (const float*)d_in[i]; };

    const float* feats  = in(0);   // (1,3,384,112,112)
    const float* points = in(1);   // (1,3,16,2)

    // JAX pytree flatten order (dict keys sorted alphabetically, lists in order)
    int ix = 2;
    const float* dec_ffn_l1_b = in(ix++); const float* dec_ffn_l1_w = in(ix++);
    const float* dec_ffn_l2_b = in(ix++); const float* dec_ffn_l2_w = in(ix++);
    const float* msda_aw_b  = in(ix++); const float* msda_aw_w  = in(ix++);
    const float* msda_off_b = in(ix++); const float* msda_off_w = in(ix++);
    const float* msda_out_b = in(ix++); const float* msda_out_w = in(ix++);
    const float* msda_val_b = in(ix++); const float* msda_val_w = in(ix++);
    MHAp decSelf;
    decSelf.bk = in(ix++); decSelf.bo = in(ix++); decSelf.bq = in(ix++); decSelf.bv = in(ix++);
    decSelf.wk = in(ix++); decSelf.wo = in(ix++); decSelf.wq = in(ix++); decSelf.wv = in(ix++);
    const float* dq_b0 = in(ix++); const float* dq_w0 = in(ix++);
    const float* dq_b1 = in(ix++); const float* dq_w1 = in(ix++);
    const float* dq_b2 = in(ix++); const float* dq_w2 = in(ix++);
    const float* dxy_b0 = in(ix++); const float* dxy_w0 = in(ix++);
    const float* dxy_b1 = in(ix++); const float* dxy_w1 = in(ix++);
    const float* dxy_b2 = in(ix++); const float* dxy_w2 = in(ix++);
    MHAp encAttn;
    encAttn.bk = in(ix++); encAttn.bo = in(ix++); encAttn.bq = in(ix++); encAttn.bv = in(ix++);
    encAttn.wk = in(ix++); encAttn.wo = in(ix++); encAttn.wq = in(ix++); encAttn.wv = in(ix++);
    const float* enc_ffn_l1_b = in(ix++); const float* enc_ffn_l1_w = in(ix++);
    const float* enc_ffn_l2_b = in(ix++); const float* enc_ffn_l2_w = in(ix++);
    const float* featup_b = in(ix++); const float* featup_w = in(ix++);
    const float* level_pos = in(ix++);
    MHAp odecCross;
    odecCross.bk = in(ix++); odecCross.bo = in(ix++); odecCross.bq = in(ix++); odecCross.bv = in(ix++);
    odecCross.wk = in(ix++); odecCross.wo = in(ix++); odecCross.wq = in(ix++); odecCross.wv = in(ix++);
    const float* odec_ffn_l1_b = in(ix++); const float* odec_ffn_l1_w = in(ix++);
    const float* odec_ffn_l2_b = in(ix++); const float* odec_ffn_l2_w = in(ix++);
    MHAp odecSelf;
    odecSelf.bk = in(ix++); odecSelf.bo = in(ix++); odecSelf.bq = in(ix++); odecSelf.bv = in(ix++);
    odecSelf.wk = in(ix++); odecSelf.wo = in(ix++); odecSelf.wq = in(ix++); odecSelf.wv = in(ix++);
    const float* query_embed = in(ix++);

    // ---- workspace carve-out (floats) ----
    float* W = (float*)d_ws;
    size_t off = 0;
    auto alloc = [&](size_t n) { float* p = W + off; off += n; return p; };
    float* featIn = alloc((size_t)2 * 12544 * 384);
    float* xbuf   = alloc((size_t)2 * 12544 * 256);
    float* srcF   = alloc((size_t)1 * 29008 * 256);
    float* srcO   = alloc((size_t)2 * 29008 * 256);
    float* valF   = alloc((size_t)1 * 29008 * 256);
    float* valO   = alloc((size_t)2 * 29008 * 256);
    float* SB[12];
    for (int i = 0; i < 12; ++i) SB[i] = alloc(65536);
    float* mem     = alloc(16 * 256);
    float* pf      = alloc(2 * 16 * 256);
    float* curq    = alloc(2 * 16 * 256);
    float* curq2   = alloc(2 * 16 * 256);
    float* qbuf    = alloc(2 * 16 * 256);
    float* ptsS    = alloc(3 * 16 * 2);
    float* otherP  = alloc(2 * 16 * 2);
    float* otherP2 = alloc(2 * 16 * 2);

    // ---- host-side helpers ----
    auto gemm = [&](const float* A, const float* B, const float* bias, float* C,
                    int M, int N, int K, int act) {
        dim3 g((N + GTN - 1) / GTN, (M + GTM - 1) / GTM);
        gemm_bias_act_wmma_kernel<<<g, 256, 0, stream>>>(A, B, bias, C, M, N, K, act);
    };
    auto ln = [&](const float* x, const float* res, float* out, int R) {
        ln_residual_kernel<<<R, 256, 0, stream>>>(x, res, out, R);
    };
    auto mha = [&](const MHAp& p, const float* qin, const float* kvin,
                   int B, int Lq, int Lk, float* out) {
        gemm(qin,  p.wq, p.bq, SB[0], B * Lq, 256, 256, 0);
        gemm(kvin, p.wk, p.bk, SB[1], B * Lk, 256, 256, 0);
        gemm(kvin, p.wv, p.bv, SB[2], B * Lk, 256, 256, 0);
        attn_kernel<<<B * 8 * Lq, 32, 0, stream>>>(SB[0], SB[1], SB[2], SB[3], Lq, Lk);
        gemm(SB[3], p.wo, p.bo, out, B * Lq, 256, 256, 0);
    };
    auto mlp3 = [&](const float* w0, const float* b0, const float* w1, const float* b1,
                    const float* w2, const float* b2, const float* x, float* out,
                    int M, int Nlast) {
        gemm(x, w0, b0, SB[10], M, 256, 256, 1);
        gemm(SB[10], w1, b1, SB[11], M, 256, 256, 1);
        gemm(SB[11], w2, b2, out, M, Nlast, 256, 0);
    };
    auto dec_layer = [&](float* tgt, const float* ref, const float* value, int Bn, int Q) {
        int M = Bn * Q;
        mha(decSelf, tgt, tgt, Bn, Q, Q, SB[4]);
        ln(SB[4], tgt, tgt, M);
        gemm(tgt, msda_off_w, msda_off_b, SB[5], M, 576, 256, 0);
        gemm(tgt, msda_aw_w,  msda_aw_b,  SB[6], M, 288, 256, 0);
        aw_softmax_kernel<<<(M * 8 + 63) / 64, 64, 0, stream>>>(SB[6], M);
        msda_sample_kernel<<<Bn * Q * 8, 32, 0, stream>>>(value, SB[5], SB[6], ref, SB[7], Q);
        gemm(SB[7], msda_out_w, msda_out_b, SB[4], M, 256, 256, 0);
        ln(SB[4], tgt, tgt, M);
        gemm(tgt, dec_ffn_l1_w, dec_ffn_l1_b, SB[8], M, 1024, 256, 1);
        gemm(SB[8], dec_ffn_l2_w, dec_ffn_l2_b, SB[4], M, 256, 1024, 0);
        ln(SB[4], tgt, tgt, M);
    };
    auto enc_layer = [&](float* x) {   // (1,16,256)
        mha(encAttn, x, x, 1, 16, 16, SB[4]);
        ln(SB[4], x, x, 16);
        gemm(x, enc_ffn_l1_w, enc_ffn_l1_b, SB[8], 16, 1024, 256, 1);
        gemm(SB[8], enc_ffn_l2_w, enc_ffn_l2_b, SB[4], 16, 256, 1024, 0);
        ln(SB[4], x, x, 16);
    };
    auto odec_layer = [&](float* x, int L) {   // (1,L,256), mem (1,16,256)
        mha(odecSelf, x, x, 1, L, L, SB[4]);
        ln(SB[4], x, x, L);
        mha(odecCross, x, mem, 1, L, 16, SB[4]);
        ln(SB[4], x, x, L);
        gemm(x, odec_ffn_l1_w, odec_ffn_l1_b, SB[8], L, 1024, 256, 1);
        gemm(SB[8], odec_ffn_l2_w, odec_ffn_l2_b, SB[4], L, 256, 1024, 0);
        ln(SB[4], x, x, L);
    };
    auto get_srcs = [&](const float* framesBase, int Nimg, float* src) {
        size_t n1 = (size_t)Nimg * 384 * 12544;
        chw2hwc_kernel<<<(unsigned)((n1 + 255) / 256), 256, 0, stream>>>(framesBase, featIn, Nimg);
        gemm(featIn, featup_w, featup_b, xbuf, Nimg * 12544, 256, 384, 0);
        const int Hs[4] = {28, 56, 112, 112};
        const int st[4] = {0, 784, 3920, 16464};
        for (int l = 0; l < 4; ++l) {
            size_t tot = (size_t)Nimg * Hs[l] * Hs[l] * 256;
            resize_to_src_kernel<<<(unsigned)((tot + 255) / 256), 256, 0, stream>>>(
                xbuf, src, Nimg, Hs[l], st[l]);
        }
        ln(src, nullptr, src, Nimg * 29008);
        size_t tot = (size_t)Nimg * 29008 * 256;
        add_pos2d_kernel<<<(unsigned)((tot + 255) / 256), 256, 0, stream>>>(src, level_pos, Nimg);
    };

    // ================= pipeline =================
    scale_copy_kernel<<<1, 256, 0, stream>>>(points, ptsS, 96, 1.f / 224.f);

    // first frame: srcs + value, decode initial queries, build memory
    get_srcs(feats, 1, srcF);
    gemm(srcF, msda_val_w, msda_val_b, valF, 29008, 256, 256, 0);
    bcast_rows_kernel<<<(16 * 256 + 255) / 256, 256, 0, stream>>>(query_embed, pf, 16 * 256, 16);
    dec_layer(pf, ptsS, valF, 1, 16);
    ln(pf, nullptr, pf, 16);
    add_pos1d_kernel<<<(16 * 256 + 255) / 256, 256, 0, stream>>>(pf, 16 * 256, 16);
    enc_layer(pf);
    hipMemcpyAsync(mem, pf, 16 * 256 * sizeof(float), hipMemcpyDeviceToDevice, stream);

    // other frames: srcs + value (hoisted out of the refinement loop)
    get_srcs(feats + (size_t)384 * 12544, 2, srcO);
    gemm(srcO, msda_val_w, msda_val_b, valO, 2 * 29008, 256, 256, 0);
    extreme4_kernel<<<1, 1, 0, stream>>>(ptsS, otherP);
    bcast_rows_kernel<<<(2 * 4 * 256 + 255) / 256, 256, 0, stream>>>(query_embed, curq, 2 * 4 * 256, 4);

    float* out_f = (float*)d_out;
    int cur_p = 4;
    int outBase = 0;
    for (int it = 0; it <= 2; ++it) {
        int slice = 2 * cur_p * 2;          // (F-1)*cur_p*2 floats per refinement
        for (int r = 0; r < 2; ++r) {
            hipMemcpyAsync(pf, curq, (size_t)2 * cur_p * 256 * sizeof(float),
                           hipMemcpyDeviceToDevice, stream);
            dec_layer(pf, otherP, valO, 2, cur_p);
            int L = 2 * cur_p;
            ln(pf, nullptr, qbuf, L);
            add_pos1d_kernel<<<(L * 256 + 255) / 256, 256, 0, stream>>>(qbuf, L * 256, L);
            odec_layer(qbuf, L);
            mlp3(dxy_w0, dxy_b0, dxy_w1, dxy_b1, dxy_w2, dxy_b2, qbuf, SB[9], L, 2);
            update_points_kernel<<<1, 256, 0, stream>>>(otherP, SB[9],
                                                        out_f + outBase + r * slice, slice);
            mlp3(dq_w0, dq_b0, dq_w1, dq_b1, dq_w2, dq_b2, pf, SB[9], L, 256);
            axpy_kernel<<<(L * 256 + 255) / 256, 256, 0, stream>>>(curq, SB[9], L * 256);
        }
        outBase += 2 * slice;
        if (it == 2) break;
        subdivide_points_kernel<<<1, 256, 0, stream>>>(otherP, otherP2, cur_p);
        { float* t = otherP; otherP = otherP2; otherP2 = t; }
        interleave_q_kernel<<<(2 * 2 * cur_p * 256 + 255) / 256, 256, 0, stream>>>(
            query_embed, curq, curq2, cur_p);
        { float* t = curq; curq = curq2; curq2 = t; }
        cur_p *= 2;
    }
}
